// RNN_OSC_71201967833342
// MI455X (gfx1250) — matrix operand order repaired and checked by reading, not compiled, not run on hardware
//
#include <hip/hip_runtime.h>
#include <hip/hip_bf16.h>

typedef __attribute__((ext_vector_type(16))) __bf16       v16bf;
typedef __attribute__((ext_vector_type(2)))  __bf16       v2bf;
typedef __attribute__((ext_vector_type(8)))  float        v8f;
typedef __attribute__((ext_vector_type(2)))  float        v2f;
typedef __attribute__((ext_vector_type(4)))  unsigned int u32x4;
typedef __attribute__((ext_vector_type(8)))  int          i32x8;
typedef __attribute__((ext_vector_type(4)))  int          i32x4;

#define Bsz 1024
#define Tsz 2048
#define Dsz 512
#define Usz 256
#define ROWS 16                     // batch rows per scan block

// native bf16 conversions (single v_cvt ops on gfx1250, RNE)
__device__ __forceinline__ unsigned short f2bf(float f) {
  __bf16 b = (__bf16)f;
  return __builtin_bit_cast(unsigned short, b);
}
__device__ __forceinline__ unsigned pkbf(float lo, float hi) {
  v2f t; t.x = lo; t.y = hi;
  v2bf r = __builtin_convertvector(t, v2bf);     // v_cvt_pk_bf16_f32
  return __builtin_bit_cast(unsigned, r);
}
__device__ __forceinline__ float bf2f(unsigned short h) {
  unsigned u = ((unsigned)h) << 16;
  return __builtin_bit_cast(float, u);
}

union Frag { v16bf v; unsigned short s[16]; unsigned u[8]; };

// Byte offset of h element (m, kg) in the bf16 staging buffer laid out exactly
// as WMMA 16x32 bf16 A-fragments, k = 16*half + e (contiguous, not the ISA table order). m in [0,16), kg in [0,256).
// Lane = (m&15) | (kl[4]<<4), element e = kl[3:0], fragment = kg>>5.
__device__ __forceinline__ int a_off(int m, int kg) {
  int c  = kg >> 5;
  int kl = kg & 31;
  int ha = (kl >> 4) & 1;
  int e  = kl & 15;
  int la = (m & 15) | (ha << 4);
  return (c << 10) + (la << 5) + (e << 1);
}

// ---------------------------------------------------------------------------
// Phase 1: seqT[d][b] = bf16( sum_t x[b][t] * Wl[t][d] + bl[d] )
// GEMM M=B(1024) x N=D(512) x K=T(2048), bf16 WMMA, f32 accumulate.
// Each wave: 4 m-tiles x 1 n-tile (reuses the strided Wl fragment 4x).
// ---------------------------------------------------------------------------
__global__ __launch_bounds__(256) void seq_gemm_kernel(
    const float* __restrict__ x, const float* __restrict__ Wl,
    const float* __restrict__ bl, unsigned short* __restrict__ seqT) {
  const int lane = threadIdx.x & 31;
  const int g    = blockIdx.x * 8 + (threadIdx.x >> 5);   // 512 wave-groups
  const int n0   = (g & 31) << 4;                         // d-tile origin
  const int m0b  = (g >> 5) << 6;                         // 4 m-tiles (b rows)
  const int hlf  = lane >> 4, nin = lane & 15;

  v8f acc[4] = {};

  for (int kb = 0; kb < Tsz; kb += 32) {
    Frag bfr;
    const int krow = kb + (hlf ? 16 : 0);
    const int col  = n0 + nin;
#pragma unroll
    for (int e = 0; e < 16; e += 2)
      bfr.u[e >> 1] = pkbf(Wl[(size_t)(krow + e)     * Dsz + col],
                           Wl[(size_t)(krow + e + 1) * Dsz + col]);
#pragma unroll
    for (int i = 0; i < 4; ++i) {
      Frag af;
      const float* xr = x + (size_t)(m0b + (i << 4) + nin) * Tsz + kb + 16 * hlf;
#pragma unroll
      for (int j = 0; j < 4; ++j) af.u[j]     = pkbf(xr[2 * j],      xr[2 * j + 1]);      // K=16h+e
#pragma unroll
      for (int j = 4; j < 8; ++j) af.u[j]     = pkbf(xr[2 * j],      xr[2 * j + 1]);      // K=16h+e
      acc[i] = __builtin_amdgcn_wmma_f32_16x16x32_bf16(
          false, af.v, false, bfr.v, (short)0, acc[i], false, false);
    }
  }

  const int   d   = n0 + nin;
  const float blv = bl[d];
#pragma unroll
  for (int i = 0; i < 4; ++i) {
    unsigned pk[4];
#pragma unroll
    for (int j = 0; j < 4; ++j)
      pk[j] = pkbf(acc[i][2 * j] + blv, acc[i][2 * j + 1] + blv);
    const int mrow = m0b + (i << 4) + 8 * hlf;           // rows mrow..mrow+7
    unsigned* dst  = (unsigned*)(seqT + (size_t)d * Bsz + mrow);
    dst[0] = pk[0]; dst[1] = pk[1]; dst[2] = pk[2]; dst[3] = pk[3];
  }
}

// ---------------------------------------------------------------------------
// Phase 2: h0 = tanh(cond@Wh+bh); 512x scan h <- seq[:,d] (x) k + h@R + br;
// FiLM + tanh(h@Wo+bo). One block = 16 batch rows, 8 waves = 1x8 C-tile row.
// R register-resident (bf16 B-frags, 128 VGPRs/wave); h ping-pongs through LDS
// in pre-swizzled A-fragment layout; seqT tile DMA'd by the Tensor Data Mover.
// ---------------------------------------------------------------------------
__global__ __launch_bounds__(256) void rnn_scan_kernel(
    const unsigned short* __restrict__ seqT,
    const float* __restrict__ cond, const float* __restrict__ kvec,
    const float* __restrict__ R,    const float* __restrict__ br,
    const float* __restrict__ Wh,   const float* __restrict__ bh,
    const float* __restrict__ Wf,   const float* __restrict__ bfv,
    const float* __restrict__ Wo,   const float* __restrict__ bo,
    float* __restrict__ out) {
  __shared__ __align__(16) unsigned short seq_lds[Dsz * ROWS];   // 16 KB (reused as f32 h at end)
  __shared__ __align__(16) unsigned char  hbuf[2][8192];         // 2 x 8 KB ping-pong

  const int tid  = threadIdx.x;
  const int lane = tid & 31;
  const int wc   = tid >> 5;     // wave = 32-wide column group == next-step k-chunk
  const int hlf  = lane >> 4;
  const int nin  = lane & 15;
  const int bt   = blockIdx.x;

  // ---- async fetch of seqT[:, 16*bt .. 16*bt+16) (bf16) into LDS via TDM ----
#if __has_builtin(__builtin_amdgcn_tensor_load_to_lds) && __has_builtin(__builtin_amdgcn_s_wait_tensorcnt)
  if (wc == 0) {
    unsigned lds_off       = (unsigned)(unsigned long long)(void*)&seq_lds[0];
    unsigned long long ga  = (unsigned long long)(const void*)(seqT + (size_t)bt * ROWS);
    u32x4 g0 = { 1u,                                  // count=1, no gather
                 lds_off,
                 (unsigned)(ga & 0xFFFFFFFFu),
                 (unsigned)((ga >> 32) & 0x1FFFFFFu) | (2u << 30) };  // type=2
    i32x8 g1 = { (int)(1u << 16),                     // data_size = 2 bytes
                 (int)(((unsigned)Bsz & 0xFFFFu) << 16),              // tensor_dim0 lo
                 (int)(((unsigned)Bsz >> 16) | ((unsigned)Dsz << 16)),// dim0 hi | dim1 lo
                 (int)((unsigned)ROWS << 16),         // dim1 hi=0 | tile_dim0=16
                 (int)Dsz,                            // tile_dim1=512 | tile_dim2=0
                 (int)Bsz,                            // tensor_dim0_stride (elems)
                 0, 0 };
    i32x4 gz  = {0, 0, 0, 0};
    i32x8 gz8 = {0, 0, 0, 0, 0, 0, 0, 0};
    __builtin_amdgcn_tensor_load_to_lds(g0, g1, gz, gz, gz8, 0);
  }
#else
  for (int i = tid; i < Dsz * (ROWS / 2); i += 256) {
    int d = i >> 3, j = i & 7;
    ((unsigned*)seq_lds)[(d << 3) + j] =
        ((const unsigned*)(seqT + (size_t)d * Bsz + bt * ROWS))[j];
  }
#endif

  // ---- preload R as register-resident bf16 B-fragments (2 n-tiles x 8 k) ----
  v16bf Bf[16];
  {
    const int ncol = wc << 5;
#pragma unroll
    for (int c = 0; c < 8; ++c)
#pragma unroll
      for (int t = 0; t < 2; ++t) {
        Frag f;
        const int kb = (c << 5) + ((lane < 16) ? 0 : 16);
        const int n  = ncol + (t << 4) + nin;
#pragma unroll
        for (int e = 0; e < 16; e += 2)
          f.u[e >> 1] = pkbf(R[(size_t)(kb + e) * Usz + n],
                             R[(size_t)(kb + e + 1) * Usz + n]);
        Bf[(c << 1) + t] = f.v;
      }
  }

  // ---- h0 = tanh(cond @ Wh + bh), written pre-swizzled into hbuf[0] ----
  {
    const int m = tid >> 4, kq = tid & 15, row = bt * ROWS + m;
    const float c0 = cond[row * 4 + 0], c1 = cond[row * 4 + 1];
    const float c2 = cond[row * 4 + 2], c3 = cond[row * 4 + 3];
#pragma unroll
    for (int j = 0; j < 16; ++j) {
      const int kg = (kq << 4) + j;
      float v = bh[kg] + c0 * Wh[kg] + c1 * Wh[Usz + kg]
                       + c2 * Wh[2 * Usz + kg] + c3 * Wh[3 * Usz + kg];
      *(unsigned short*)(&hbuf[0][0] + a_off(m, kg)) = f2bf(tanhf(v));
    }
  }

#if __has_builtin(__builtin_amdgcn_tensor_load_to_lds) && __has_builtin(__builtin_amdgcn_s_wait_tensorcnt)
  if (wc == 0) __builtin_amdgcn_s_wait_tensorcnt(0);
#endif
  __syncthreads();

  const float kv0 = kvec[(wc << 5) + nin], kv1 = kvec[(wc << 5) + 16 + nin];
  const float bv0 = br  [(wc << 5) + nin], bv1 = br  [(wc << 5) + 16 + nin];
  const int soff0 = a_off(8 * hlf, (wc << 5) + nin);        // +32B per r
  const int soff1 = a_off(8 * hlf, (wc << 5) + 16 + nin);

  v8f a0 = {}, a1 = {};
  unsigned cur = 0;
#pragma unroll 1
  for (int d = 0; d < Dsz; ++d) {
    // per-row drive: 8 bf16 seq values for this lane's C rows, one b128 load
    u32x4 su = *(const u32x4*)(seq_lds + d * ROWS + 8 * hlf);
    float sv[8];
#pragma unroll
    for (int j = 0; j < 4; ++j) {
      sv[2 * j]     = bf2f((unsigned short)su[j]);
      sv[2 * j + 1] = bf2f((unsigned short)(su[j] >> 16));
    }
#pragma unroll
    for (int r = 0; r < 8; ++r) { a0[r] = sv[r] * kv0 + bv0; a1[r] = sv[r] * kv1 + bv1; }

    const unsigned char* hb = &hbuf[cur][0];
#pragma unroll
    for (int c = 0; c < 8; ++c) {
      v16bf af = *(const v16bf*)(hb + (c << 10) + (lane << 5));
      a0 = __builtin_amdgcn_wmma_f32_16x16x32_bf16(false, af, false, Bf[2 * c + 0],
                                                   (short)0, a0, false, false);
      a1 = __builtin_amdgcn_wmma_f32_16x16x32_bf16(false, af, false, Bf[2 * c + 1],
                                                   (short)0, a1, false, false);
    }

    // round to bf16 (packed cvt), store swizzled as next step's A-fragments
    unsigned char* hn = &hbuf[cur ^ 1][0];
#pragma unroll
    for (int j = 0; j < 4; ++j) {
      unsigned p0 = pkbf(a0[2 * j], a0[2 * j + 1]);
      unsigned p1 = pkbf(a1[2 * j], a1[2 * j + 1]);
      *(unsigned short*)(hn + soff0 + ((2 * j)     << 5)) = (unsigned short)p0;
      *(unsigned short*)(hn + soff0 + ((2 * j + 1) << 5)) = (unsigned short)(p0 >> 16);
      *(unsigned short*)(hn + soff1 + ((2 * j)     << 5)) = (unsigned short)p1;
      *(unsigned short*)(hn + soff1 + ((2 * j + 1) << 5)) = (unsigned short)(p1 >> 16);
    }
    cur ^= 1;
    __syncthreads();
  }

  // ---- stash final h (full f32) over the (now dead) seq tile ----
  {
    float* h32 = (float*)seq_lds;                     // 16 x 256 f32 = 16 KB
#pragma unroll
    for (int r = 0; r < 8; ++r) {
      const int m = r + 8 * hlf;
      h32[m * Usz + (wc << 5) + nin]      = a0[r];
      h32[m * Usz + (wc << 5) + 16 + nin] = a1[r];
    }
  }
  __syncthreads();

  // ---- FiLM + output head: out = tanh((h*gamma+beta) @ Wo + bo) ----
  if (tid < ROWS) {
    const int row = bt * ROWS + tid;
    const float c0 = cond[row * 4 + 0], c1 = cond[row * 4 + 1];
    const float c2 = cond[row * 4 + 2], c3 = cond[row * 4 + 3];
    const float* h32 = (const float*)seq_lds;
    float s = 0.f;
    for (int n = 0; n < Usz; ++n) {
      float g  = bfv[n]       + c0 * Wf[n]        + c1 * Wf[512 + n]
                              + c2 * Wf[1024 + n] + c3 * Wf[1536 + n];
      float be = bfv[256 + n] + c0 * Wf[256 + n]  + c1 * Wf[768 + n]
                              + c2 * Wf[1280 + n] + c3 * Wf[1792 + n];
      s += (h32[tid * Usz + n] * g + be) * Wo[n];
    }
    out[row] = tanhf(s + bo[0]);
  }
}

extern "C" void kernel_launch(void* const* d_in, const int* in_sizes, int n_in,
                              void* d_out, int out_size, void* d_ws, size_t ws_size,
                              hipStream_t stream) {
  (void)in_sizes; (void)n_in; (void)out_size; (void)ws_size;
  const float* x    = (const float*)d_in[0];
  const float* cond = (const float*)d_in[1];
  const float* Wl   = (const float*)d_in[2];
  const float* bl   = (const float*)d_in[3];
  const float* k    = (const float*)d_in[4];
  const float* R    = (const float*)d_in[5];
  const float* br   = (const float*)d_in[6];
  const float* Wh   = (const float*)d_in[7];
  const float* bh   = (const float*)d_in[8];
  const float* Wf   = (const float*)d_in[9];
  const float* bf   = (const float*)d_in[10];
  const float* Wo   = (const float*)d_in[11];
  const float* bo   = (const float*)d_in[12];

  unsigned short* seqT = (unsigned short*)d_ws;   // 512*1024 bf16 = 1 MB scratch

  seq_gemm_kernel<<<64, 256, 0, stream>>>(x, Wl, bl, seqT);
  rnn_scan_kernel<<<Bsz / ROWS, 256, 0, stream>>>(seqT, cond, k, R, br, Wh, bh,
                                                  Wf, bf, Wo, bo, (float*)d_out);
}
